// WatsonCrickMultiHeadedAttention_37005438223140
// MI455X (gfx1250) — matrix operand order repaired and checked by reading, not compiled, not run on hardware
//
#include <hip/hip_runtime.h>

#define BSZ 8
#define SEQ 1024
#define DM  512
#define NH  8
#define DK  64

typedef __attribute__((ext_vector_type(16))) _Float16 v16h;
typedef __attribute__((ext_vector_type(8)))  float    v8f;
typedef __attribute__((ext_vector_type(4)))  float    v4f;
typedef __attribute__((ext_vector_type(4)))  _Float16 v4h;
typedef __attribute__((ext_vector_type(2)))  _Float16 h2;

static __device__ __forceinline__ v8f wmma_f16(v16h a, v16h b, v8f c) {
  // (neg_a, A, neg_b, B, c_mod, C, reuse_a, reuse_b)
  return __builtin_amdgcn_wmma_f32_16x16x32_f16(false, a, false, b, (short)0, c, false, false);
}

static __device__ __forceinline__ v8f vzero8() {
  return (v8f){0.f, 0.f, 0.f, 0.f, 0.f, 0.f, 0.f, 0.f};
}

// ---- fragment loaders (f16 sources; conversion done in a pre-pass) -------
// A fragment (16x32 f16), source row-major [row][k], leading dim ld (elems).
// lanes 0-15 rows 0-15; VGPR v<4 -> K = gr*8 + (v&3)*2 + h,
// VGPR v>=4 -> K = 16 + gr*8 + ((v-4)&3)*2 + h   (gr = lane/16)
static __device__ __forceinline__ v16h load_a_f16p(const _Float16* p, int ld, int koff) {
  const int lane = threadIdx.x & 31;
  const int m = lane & 15, gr = lane >> 4;
  v16h a;
#pragma unroll
  for (int v = 0; v < 8; ++v) {
    int k = koff + ((v < 4) ? 0 : 16) + gr * 8 + (v & 3) * 2;
    h2 pr = *(const h2*)(p + m * ld + k);
    a[2 * v] = pr.x; a[2 * v + 1] = pr.y;
  }
  return a;
}

// B fragment (32x16 f16), source holds B[k][n] as row-major [n][k] (k contiguous).
// lane n = lane&15, VGPR v<4 -> K = gr*8 + (v&3)*2 + h, VGPR v>=4 -> K = 16 + gr*8 + ((v-4)&3)*2 + h
static __device__ __forceinline__ v16h load_b_f16p(const _Float16* p, int ld, int koff) {
  const int lane = threadIdx.x & 31;
  const int n = lane & 15, gr = lane >> 4;
  v16h b;
#pragma unroll
  for (int v = 0; v < 8; ++v) {
    int k = koff + ((v < 4) ? 0 : 16) + gr * 8 + (v & 3) * 2;
    h2 pr = *(const h2*)(p + n * ld + k);
    b[2 * v] = pr.x; b[2 * v + 1] = pr.y;
  }
  return b;
}

// ---- kernel A: bulk f32 -> f16 conversion (one streaming pass) -----------
__global__ void cvt16_kernel(const float* __restrict__ in, _Float16* __restrict__ out, int n) {
  int i = (blockIdx.x * blockDim.x + threadIdx.x) * 8;
  if (i >= n) return;
  v4f a = *(const v4f*)(in + i);
  v4f b = *(const v4f*)(in + i + 4);
  v4h ha, hb;
#pragma unroll
  for (int j = 0; j < 4; ++j) { ha[j] = (_Float16)a[j]; hb[j] = (_Float16)b[j]; }
  *(v4h*)(out + i) = ha;
  *(v4h*)(out + i + 4) = hb;
}

// ---- kernel B: per-row softmax stats of wc_matrix ------------------------
__global__ void wcstats_kernel(const float* __restrict__ wc,
                               float* __restrict__ wmax, float* __restrict__ wsum) {
  const int wave = threadIdx.x >> 5;
  const int lane = threadIdx.x & 31;
  const int row = blockIdx.x * 8 + wave;                 // 0 .. BSZ*SEQ-1
  const float* r = wc + (size_t)row * SEQ;
  float mx = -1e30f;
  for (int i = lane; i < SEQ; i += 32) mx = fmaxf(mx, r[i]);
#pragma unroll
  for (int m = 1; m < 32; m <<= 1) mx = fmaxf(mx, __shfl_xor(mx, m, 32));
  float s = 0.f;
  for (int i = lane; i < SEQ; i += 32) s += __expf(r[i] - mx);
#pragma unroll
  for (int m = 1; m < 32; m <<= 1) s += __shfl_xor(s, m, 32);
  if (lane == 0) { wmax[row] = mx; wsum[row] = s; }
}

// ---- kernel C: projection  out = X @ W^T + b  (f16 in, f16 out) ----------
// Unrolled ping-pong pipeline: loads target loop-carried registers directly,
// no register-rotation copies; WMMAs overlap the next block's loads.
// mode 0: out[((b*NH+h)*SEQ + n)*DK + d]   (q / k head layout)
// mode 1: out[((b*NH+h)*DK + d)*SEQ + n]   (v transposed for PV B-frags)
__global__ void proj16_kernel(const _Float16* __restrict__ X, const _Float16* __restrict__ W,
                              const float* __restrict__ bias, _Float16* __restrict__ out,
                              int mode) {
  const int rb = blockIdx.x;          // 16-row block of X (0..511)
  const int cg = blockIdx.y;          // 64-col group == head (0..7)
  const int lane = threadIdx.x & 31;
  const int n16 = lane & 15, gr = lane >> 4;

  v8f acc[4];
#pragma unroll
  for (int f = 0; f < 4; ++f) acc[f] = vzero8();

  const _Float16* Xb = X + (size_t)rb * 16 * DM;
  const _Float16* Wb = W + (size_t)cg * 64 * DM;

  v16h a0 = load_a_f16p(Xb, DM, 0);
  v16h b0[4];
#pragma unroll
  for (int f = 0; f < 4; ++f) b0[f] = load_b_f16p(Wb + (size_t)f * 16 * DM, DM, 0);

#pragma unroll
  for (int k0 = 0; k0 < DM; k0 += 64) {
    v16h a1 = load_a_f16p(Xb, DM, k0 + 32);
    v16h b1[4];
#pragma unroll
    for (int f = 0; f < 4; ++f) b1[f] = load_b_f16p(Wb + (size_t)f * 16 * DM, DM, k0 + 32);
#pragma unroll
    for (int f = 0; f < 4; ++f) acc[f] = wmma_f16(a0, b0[f], acc[f]);
    if (k0 + 64 < DM) {
      a0 = load_a_f16p(Xb, DM, k0 + 64);
#pragma unroll
      for (int f = 0; f < 4; ++f) b0[f] = load_b_f16p(Wb + (size_t)f * 16 * DM, DM, k0 + 64);
    }
#pragma unroll
    for (int f = 0; f < 4; ++f) acc[f] = wmma_f16(a1, b1[f], acc[f]);
  }

#pragma unroll
  for (int f = 0; f < 4; ++f) {
#pragma unroll
    for (int v = 0; v < 8; ++v) {
      int row = rb * 16 + v + 8 * gr;            // bn
      int j   = cg * 64 + f * 16 + n16;          // output column
      float val = acc[f][v] + bias[j];
      int b = row >> 10, n = row & (SEQ - 1);
      int h = cg, d = j & 63;
      size_t idx = (mode == 0)
          ? (((size_t)(b * NH + h) * SEQ + n) * DK + d)
          : (((size_t)(b * NH + h) * DK + d) * SEQ + n);
      out[idx] = (_Float16)val;
    }
  }
}

// ---- kernel D: fused flash attention with WC modulation (32 rows / wave) -
__global__ void attn_kernel(const _Float16* __restrict__ qh, const _Float16* __restrict__ kh,
                            const _Float16* __restrict__ vT, const float* __restrict__ wc,
                            const float* __restrict__ wcmax, const float* __restrict__ wcsum,
                            _Float16* __restrict__ aout) {
  const int rb = blockIdx.x;           // 32-row block (0..31)
  const int bh = blockIdx.y;           // b*NH + h    (0..63)
  const int b  = bh >> 3;
  const int lane = threadIdx.x & 31;
  const int n16 = lane & 15, gr = lane >> 4;

  const _Float16* q  = qh + (size_t)bh * SEQ * DK;
  const _Float16* kk = kh + (size_t)bh * SEQ * DK;
  const _Float16* vt = vT + (size_t)bh * DK * SEQ;
  const float* wcb = wc + (size_t)b * SEQ * SEQ;

  // Q fragments for both 16-row sub-tiles (kept in registers)
  v16h qa[2][2];
#pragma unroll
  for (int r = 0; r < 2; ++r) {
    const _Float16* qbase = q + (size_t)(rb * 32 + r * 16) * DK;
    qa[r][0] = load_a_f16p(qbase, DK, 0);
    qa[r][1] = load_a_f16p(qbase, DK, 32);
  }

  // per-row wc softmax stats (lane owns rows v + 8*gr of each sub-tile)
  float wmaxv[2][8], winv[2][8];
#pragma unroll
  for (int r = 0; r < 2; ++r)
#pragma unroll
    for (int v = 0; v < 8; ++v) {
      int row = rb * 32 + r * 16 + v + 8 * gr;
      wmaxv[r][v] = wcmax[b * SEQ + row];
      winv[r][v]  = 1.f / wcsum[b * SEQ + row];
    }

  v8f O[2][4];
  float mrun[2][8], lrun[2][8];
#pragma unroll
  for (int r = 0; r < 2; ++r) {
#pragma unroll
    for (int f = 0; f < 4; ++f) O[r][f] = vzero8();
#pragma unroll
    for (int v = 0; v < 8; ++v) { mrun[r][v] = -1e30f; lrun[r][v] = 0.f; }
  }

  __shared__ _Float16 pb[2][16 * 32];  // P tiles: f32 D-layout -> f16 A-layout

  for (int j0 = 0; j0 < SEQ; j0 += 32) {
    // prefetch next K / V^T blocks into cache (global_prefetch_b8)
    if (j0 + 32 < SEQ) {
      __builtin_prefetch((const void*)(kk + (size_t)(j0 + 32 + lane) * DK), 0, 3);
      __builtin_prefetch((const void*)(vt + (size_t)lane * SEQ + j0 + 32), 0, 3);
      __builtin_prefetch((const void*)(vt + (size_t)(lane + 32) * SEQ + j0 + 32), 0, 3);
    }

    // --- load all K fragments for this 32-key block, then 8 WMMAs --------
    v16h kb[4];
    kb[0] = load_b_f16p(kk + (size_t)j0 * DK, DK, 0);
    kb[1] = load_b_f16p(kk + (size_t)j0 * DK, DK, 32);
    kb[2] = load_b_f16p(kk + (size_t)(j0 + 16) * DK, DK, 0);
    kb[3] = load_b_f16p(kk + (size_t)(j0 + 16) * DK, DK, 32);

    v8f s[2][2];
#pragma unroll
    for (int t = 0; t < 2; ++t)
#pragma unroll
      for (int r = 0; r < 2; ++r) {
        v8f acc = vzero8();
        acc = wmma_f16(qa[r][0], kb[2 * t], acc);
        acc = wmma_f16(qa[r][1], kb[2 * t + 1], acc);
        s[t][r] = acc;
      }

    float pv[2][16];
#pragma unroll
    for (int t = 0; t < 2; ++t)
#pragma unroll
      for (int r = 0; r < 2; ++r)
#pragma unroll
        for (int v = 0; v < 8; ++v) {
          int row = rb * 32 + r * 16 + v + 8 * gr;
          int col = j0 + t * 16 + n16;
          float wcf = __expf(wcb[(size_t)row * SEQ + col] - wmaxv[r][v]) * winv[r][v];
          pv[r][t * 8 + v] = s[t][r][v] * 0.125f * (1.f + wcf);  // scale * (1 + wc_sm)
        }

#pragma unroll
    for (int r = 0; r < 2; ++r) {
      // row-wise max over the 32 columns (cross-lane within 16-lane group)
      float cmax[8];
#pragma unroll
      for (int v = 0; v < 8; ++v) cmax[v] = fmaxf(pv[r][v], pv[r][8 + v]);
#pragma unroll
      for (int m = 1; m < 16; m <<= 1)
#pragma unroll
        for (int v = 0; v < 8; ++v) cmax[v] = fmaxf(cmax[v], __shfl_xor(cmax[v], m, 32));

      float alpha[8], rsum[8];
#pragma unroll
      for (int v = 0; v < 8; ++v) {
        float mn = fmaxf(mrun[r][v], cmax[v]);
        alpha[v] = __expf(mrun[r][v] - mn);
        mrun[r][v] = mn;
        pv[r][v]     = __expf(pv[r][v]     - mn);
        pv[r][8 + v] = __expf(pv[r][8 + v] - mn);
        rsum[v] = pv[r][v] + pv[r][8 + v];
      }
#pragma unroll
      for (int m = 1; m < 16; m <<= 1)
#pragma unroll
        for (int v = 0; v < 8; ++v) rsum[v] += __shfl_xor(rsum[v], m, 32);
#pragma unroll
      for (int v = 0; v < 8; ++v) lrun[r][v] = lrun[r][v] * alpha[v] + rsum[v];
#pragma unroll
      for (int f = 0; f < 4; ++f)
#pragma unroll
        for (int v = 0; v < 8; ++v) O[r][f][v] *= alpha[v];

      // stage P into LDS as row-major f16 [m][k]
#pragma unroll
      for (int t = 0; t < 2; ++t)
#pragma unroll
        for (int v = 0; v < 8; ++v)
          pb[r][(v + 8 * gr) * 32 + t * 16 + n16] = (_Float16)pv[r][t * 8 + v];
    }
    __syncthreads();

    // --- load all V^T fragments, P fragments, then 8 WMMAs ---------------
    v16h pa0 = load_a_f16p(&pb[0][0], 32, 0);
    v16h pa1 = load_a_f16p(&pb[1][0], 32, 0);
    v16h vb[4];
#pragma unroll
    for (int dc = 0; dc < 4; ++dc)
      vb[dc] = load_b_f16p(vt + (size_t)dc * 16 * SEQ + j0, SEQ, 0);
#pragma unroll
    for (int dc = 0; dc < 4; ++dc) {
      O[0][dc] = wmma_f16(pa0, vb[dc], O[0][dc]);
      O[1][dc] = wmma_f16(pa1, vb[dc], O[1][dc]);
    }
    __syncthreads();
  }

  // epilogue: divide by l, write merged-head f16 a[b][n][h*64+d]
#pragma unroll
  for (int r = 0; r < 2; ++r)
#pragma unroll
    for (int dc = 0; dc < 4; ++dc)
#pragma unroll
      for (int v = 0; v < 8; ++v) {
        float o = O[r][dc][v] / lrun[r][v];
        int row = rb * 32 + r * 16 + v + 8 * gr;
        int j = (bh & 7) * 64 + dc * 16 + n16;
        aout[((size_t)b * SEQ + row) * DM + j] = (_Float16)o;
      }
}

// ---- kernel E: output projection  out = A @ Wo^T + bo  (f16 in, f32 out) -
__global__ void outproj16_kernel(const _Float16* __restrict__ A, const _Float16* __restrict__ W,
                                 const float* __restrict__ bias, float* __restrict__ out) {
  const int rb = blockIdx.x;
  const int cg = blockIdx.y;
  const int lane = threadIdx.x & 31;
  const int n16 = lane & 15, gr = lane >> 4;

  v8f acc[4];
#pragma unroll
  for (int f = 0; f < 4; ++f) acc[f] = vzero8();

  const _Float16* Ab = A + (size_t)rb * 16 * DM;
  const _Float16* Wb = W + (size_t)cg * 64 * DM;

  v16h a0 = load_a_f16p(Ab, DM, 0);
  v16h b0[4];
#pragma unroll
  for (int f = 0; f < 4; ++f) b0[f] = load_b_f16p(Wb + (size_t)f * 16 * DM, DM, 0);

#pragma unroll
  for (int k0 = 0; k0 < DM; k0 += 64) {
    v16h a1 = load_a_f16p(Ab, DM, k0 + 32);
    v16h b1[4];
#pragma unroll
    for (int f = 0; f < 4; ++f) b1[f] = load_b_f16p(Wb + (size_t)f * 16 * DM, DM, k0 + 32);
#pragma unroll
    for (int f = 0; f < 4; ++f) acc[f] = wmma_f16(a0, b0[f], acc[f]);
    if (k0 + 64 < DM) {
      a0 = load_a_f16p(Ab, DM, k0 + 64);
#pragma unroll
      for (int f = 0; f < 4; ++f) b0[f] = load_b_f16p(Wb + (size_t)f * 16 * DM, DM, k0 + 64);
    }
#pragma unroll
    for (int f = 0; f < 4; ++f) acc[f] = wmma_f16(a1, b1[f], acc[f]);
  }

#pragma unroll
  for (int f = 0; f < 4; ++f) {
#pragma unroll
    for (int v = 0; v < 8; ++v) {
      int row = rb * 16 + v + 8 * gr;
      int j = cg * 64 + f * 16 + n16;
      out[(size_t)row * DM + j] = acc[f][v] + bias[j];
    }
  }
}

// ---- launcher ------------------------------------------------------------
extern "C" void kernel_launch(void* const* d_in, const int* in_sizes, int n_in,
                              void* d_out, int out_size, void* d_ws, size_t ws_size,
                              hipStream_t stream) {
  (void)in_sizes; (void)n_in; (void)out_size; (void)ws_size;
  const float* q_in = (const float*)d_in[0];
  const float* k_in = (const float*)d_in[1];
  const float* v_in = (const float*)d_in[2];
  const float* wc   = (const float*)d_in[3];
  const float* Wq = (const float*)d_in[4];
  const float* bq = (const float*)d_in[5];
  const float* Wk = (const float*)d_in[6];
  const float* bk = (const float*)d_in[7];
  const float* Wv = (const float*)d_in[8];
  const float* bv = (const float*)d_in[9];
  const float* Wo = (const float*)d_in[10];
  const float* bo = (const float*)d_in[11];
  float* out = (float*)d_out;

  const size_t MB = (size_t)1 << 20;
  char* ws = (char*)d_ws;
  _Float16* xq16 = (_Float16*)(ws);             // 8 MiB (reused as aws later)
  _Float16* xk16 = (_Float16*)(ws + 8 * MB);    // 8 MiB
  _Float16* xv16 = (_Float16*)(ws + 16 * MB);   // 8 MiB
  _Float16* qws  = (_Float16*)(ws + 24 * MB);   // 8 MiB
  _Float16* kws  = (_Float16*)(ws + 32 * MB);   // 8 MiB
  _Float16* vtws = (_Float16*)(ws + 40 * MB);   // 8 MiB
  _Float16* w16q = (_Float16*)(ws + 48 * MB);               // 512 KiB each
  _Float16* w16k = (_Float16*)(ws + 48 * MB + 512 * 1024);
  _Float16* w16v = (_Float16*)(ws + 49 * MB);
  _Float16* w16o = (_Float16*)(ws + 49 * MB + 512 * 1024);
  float* wcmax = (float*)(ws + 50 * MB);        // 32 KiB
  float* wcsum = wcmax + BSZ * SEQ;             // 32 KiB
  _Float16* aws = xq16;                         // reuse (xq16 dead after proj)

  const int NX = BSZ * SEQ * DM;   // 4,194,304
  const int NW = DM * DM;          // 262,144
  cvt16_kernel<<<dim3(NX / 2048), dim3(256), 0, stream>>>(q_in, xq16, NX);
  cvt16_kernel<<<dim3(NX / 2048), dim3(256), 0, stream>>>(k_in, xk16, NX);
  cvt16_kernel<<<dim3(NX / 2048), dim3(256), 0, stream>>>(v_in, xv16, NX);
  cvt16_kernel<<<dim3(NW / 2048), dim3(256), 0, stream>>>(Wq, w16q, NW);
  cvt16_kernel<<<dim3(NW / 2048), dim3(256), 0, stream>>>(Wk, w16k, NW);
  cvt16_kernel<<<dim3(NW / 2048), dim3(256), 0, stream>>>(Wv, w16v, NW);
  cvt16_kernel<<<dim3(NW / 2048), dim3(256), 0, stream>>>(Wo, w16o, NW);

  wcstats_kernel<<<dim3(BSZ * SEQ / 8), dim3(256), 0, stream>>>(wc, wcmax, wcsum);

  dim3 ggrid(BSZ * SEQ / 16, DM / 64);
  proj16_kernel<<<ggrid, dim3(32), 0, stream>>>(xq16, w16q, bq, qws, 0);
  proj16_kernel<<<ggrid, dim3(32), 0, stream>>>(xk16, w16k, bk, kws, 0);
  proj16_kernel<<<ggrid, dim3(32), 0, stream>>>(xv16, w16v, bv, vtws, 1);

  attn_kernel<<<dim3(SEQ / 32, BSZ * NH), dim3(32), 0, stream>>>(
      qws, kws, vtws, wc, wcmax, wcsum, aws);

  outproj16_kernel<<<ggrid, dim3(32), 0, stream>>>(aws, w16o, bo, out);
}
